// GIN_25975962206683
// MI455X (gfx1250) — matrix operand order repaired and checked by reading, not compiled, not run on hardware
//
#include <hip/hip_runtime.h>
#include <hip/hip_bf16.h>

#define NODES 100000
#define FEAT 128

typedef __attribute__((ext_vector_type(16))) _Float16 v16h;
typedef __attribute__((ext_vector_type(8)))  float    v8f;

// ---------------------------------------------------------------------------
// z = h  (prepare accumulator with self term; eps = 0 so (1+eps)h = h)
// ---------------------------------------------------------------------------
__global__ __launch_bounds__(256) void gin_copy_kernel(
    const float* __restrict__ src, float* __restrict__ dst, int n4) {
  int i = blockIdx.x * blockDim.x + threadIdx.x;
  if (i < n4) {
    ((float4*)dst)[i] = ((const float4*)src)[i];
  }
}

// ---------------------------------------------------------------------------
// z[dst[e]] += h[src[e]]   (one wave32 per edge, 4 floats per lane)
// ---------------------------------------------------------------------------
__global__ __launch_bounds__(256) void gin_scatter_kernel(
    const float* __restrict__ h, const int* __restrict__ src,
    const int* __restrict__ dst, float* __restrict__ z, int n_edges) {
  int e = blockIdx.x * (blockDim.x >> 5) + (threadIdx.x >> 5);
  if (e >= n_edges) return;
  int lane = threadIdx.x & 31;
  int s = src[e];
  int d = dst[e];
  const float4 v = *(const float4*)(h + (long)s * FEAT + lane * 4);
  float* zp = z + (long)d * FEAT + lane * 4;
  atomicAdd(zp + 0, v.x);
  atomicAdd(zp + 1, v.y);
  atomicAdd(zp + 2, v.z);
  atomicAdd(zp + 3, v.w);
}

// ---------------------------------------------------------------------------
// out = [relu](A @ W + bias)  for A: (16*mTiles) x 128, W: 128 x 128
//
// 8 waves / block, one 16-row M tile per wave. W is converted to f16 and
// stored transposed ([n][k]) in LDS so B fragments are contiguous ds loads.
//
// WMMA f16 16x16x32 per-lane layouts used here (contiguous K for A and B):
//   A: M = lane%16,  half i -> K = 32*kt + 16*(lane>=16) + i
//   B: N = lane%16,  half i -> K = 32*kt + 16*(lane>=16) + i
//   D: VGPR r -> M = r + 8*(lane>=16), N = lane%16
// ---------------------------------------------------------------------------
template <bool RELU>
__global__ __launch_bounds__(256) void gin_mlp_gemm(
    const float* __restrict__ A, const float* __restrict__ W,
    const float* __restrict__ bias, float* __restrict__ out, int mTiles) {
  __shared__ _Float16 Wt[FEAT * FEAT];  // Wt[n][k] = (f16) W[k][n]

  int tid = threadIdx.x;
  for (int idx = tid; idx < FEAT * FEAT; idx += 256) {
    int k = idx >> 7;
    int n = idx & 127;
    Wt[n * FEAT + k] = (_Float16)W[idx];
  }
  __syncthreads();

  int wave = tid >> 5;
  int lane = tid & 31;
  int mTile = blockIdx.x * 8 + wave;
  if (mTile >= mTiles) return;  // wave-uniform: EXEC stays all-ones for WMMA

  int laneLo = lane & 15;
  int laneHi = lane >> 4;  // 0 or 1
  long row = (long)mTile * 16 + laneLo;

  // Preload the four 16x32 A fragments (K = 0..127), f32 -> f16 in-register.
  v16h a[4];
  const float* ap0 = A + row * FEAT + 16 * laneHi;
#pragma unroll
  for (int kt = 0; kt < 4; ++kt) {
    const float* ap = ap0 + kt * 32;
    union { v16h v; _Float16 h[16]; } u;
#pragma unroll
    for (int i = 0; i < 8; ++i) {
      u.h[i]     = (_Float16)ap[i];       // K offsets 0..7  (+16 if upper half-wave)
      u.h[i + 8] = (_Float16)ap[i + 8];   // K offsets 8..15 (+16 if upper half-wave)
    }
    a[kt] = u.v;
  }

  // 8 N tiles of 16 columns each.
#pragma unroll
  for (int j = 0; j < 8; ++j) {
    v8f acc = {};
    const _Float16* bp = &Wt[(j * 16 + laneLo) * FEAT + 16 * laneHi];
#pragma unroll
    for (int kt = 0; kt < 4; ++kt) {
      v16h b = *(const v16h*)(bp + kt * 32);  // 32B-aligned, 2x ds_load_b128
      acc = __builtin_amdgcn_wmma_f32_16x16x32_f16(
          /*neg_a=*/false, a[kt], /*neg_b=*/false, b,
          /*c_mod=*/(short)0, acc, /*reuse_a=*/false, /*reuse_b=*/false);
    }
    float bv = bias[j * 16 + laneLo];
    float* op = out + ((long)mTile * 16 + 8 * laneHi) * FEAT + j * 16 + laneLo;
#pragma unroll
    for (int r = 0; r < 8; ++r) {
      float v = acc[r] + bv;
      if (RELU) v = fmaxf(v, 0.0f);
      op[(long)r * FEAT] = v;
    }
  }
}

// ---------------------------------------------------------------------------
// launch: 3 GIN layers, shared MLP weights
// ---------------------------------------------------------------------------
extern "C" void kernel_launch(void* const* d_in, const int* in_sizes, int n_in,
                              void* d_out, int out_size, void* d_ws, size_t ws_size,
                              hipStream_t stream) {
  const float* x   = (const float*)d_in[0];
  const int*   src = (const int*)  d_in[1];
  const int*   dst = (const int*)  d_in[2];
  const float* W1  = (const float*)d_in[3];
  const float* b1  = (const float*)d_in[4];
  const float* W2  = (const float*)d_in[5];
  const float* b2  = (const float*)d_in[6];
  const int n_edges = in_sizes[1];

  const size_t nodeFeats = (size_t)NODES * FEAT;
  float* z    = (float*)d_ws;              // (1+eps)h + aggregation
  float* hid  = z + nodeFeats;             // hidden MLP activations
  float* hbuf = hid + nodeFeats;           // inter-layer node features
  float* outp = (float*)d_out;

  const int n4         = (int)(nodeFeats / 4);
  const int copyBlocks = (n4 + 255) / 256;
  const int edgeBlocks = (n_edges + 7) / 8;  // 8 edges (waves) per block
  const int mTiles     = NODES / 16;         // 6250 exactly
  const int gemmBlocks = (mTiles + 7) / 8;

  const float* hprev = x;
  for (int layer = 0; layer < 3; ++layer) {
    gin_copy_kernel<<<copyBlocks, 256, 0, stream>>>(hprev, z, n4);
    gin_scatter_kernel<<<edgeBlocks, 256, 0, stream>>>(hprev, src, dst, z, n_edges);
    gin_mlp_gemm<true><<<gemmBlocks, 256, 0, stream>>>(z, W1, b1, hid, mTiles);
    if (layer == 2) {
      gin_mlp_gemm<false><<<gemmBlocks, 256, 0, stream>>>(hid, W2, b2, outp, mTiles);
      hprev = outp;
    } else {
      gin_mlp_gemm<true><<<gemmBlocks, 256, 0, stream>>>(hid, W2, b2, hbuf, mTiles);
      hprev = hbuf;
    }
  }
}
